// TimeTransformerNetBlock_31954556682767
// MI455X (gfx1250) — matrix operand order repaired and checked by reading, not compiled, not run on hardware
//
#include <hip/hip_runtime.h>
#include <hip/hip_bf16.h>
#include <stdint.h>

// ---------------------------------------------------------------------------
// TimeTransformerNetBlock for MI455X (gfx1250, wave32, WMMA)
//
// Algebraic restructuring (exact in infinite precision):
//   diag[c,t,h]  = x_h^T (Wq^T Wk) x_h / sqrt(E)         (Mqk precomputed)
//   vsum[c,h,:]  = (sum_t x_h) Wv^T                      (sum commutes)
//   out_pre      = y @ G + bcomb,  G[e',e] = sum_m W1[m,e']W2[e,m]
//   (no nonlinearity between fc1/fc2 -> weights compose: 1.1 TFLOP -> 137 GFLOP)
// Memory-bound after fusion (~1.5 GB HBM traffic @ 23.3 TB/s); GEMMs in bf16 WMMA.
// Main GEMM double-buffers LDS with GLOBAL_LOAD_ASYNC_TO_LDS_B128 (ASYNCcnt).
// ---------------------------------------------------------------------------

#define EMBED 1024
#define HEADS 16
#define PER   64
#define MAPD  4096
#define CDIM  32
#define TDIM  2048
#define ROWS  (CDIM*TDIM)   // 65536
#define LNEPS 1e-5f

typedef __attribute__((ext_vector_type(16))) __bf16 v16bf;
typedef __attribute__((ext_vector_type(8)))  float  v8f;
typedef __bf16 bf16;

struct alignas(16) B16x8 { uint32_t u[4]; };   // 8 packed bf16
union  FragU { v16bf v; B16x8 h[2]; };
struct alignas(16) F4  { float x, y, z, w; };
struct alignas(8)  BF4 { bf16 a, b, c, d; };

#define APITCH 56   // bf16 elems per LDS row: 112B = 16B-aligned, low bank conflict
#define BPITCH 56

// ---------------------------------------------------------------------------
// CDNA5 async global->LDS copy (tracked by ASYNCcnt).
// LDS operand: low 32 bits of the flat shared-memory pointer are the LDS
// byte address (flat LDS aperture keeps the offset in addr[31:0]).
// ---------------------------------------------------------------------------
__device__ __forceinline__ void async_load_b128(bf16* lds_ptr, const bf16* gptr)
{
  const uint32_t loff = (uint32_t)(uintptr_t)lds_ptr;
  asm volatile("global_load_async_to_lds_b128 %0, %1, off"
               :: "v"(loff), "v"(gptr) : "memory");
}
__device__ __forceinline__ void wait_asynccnt0()
{
  asm volatile("s_wait_asynccnt 0x0" ::: "memory");
}

// ---------------------------------------------------------------------------
// Shared WMMA inner step: one 32-deep K slice of a 128x128 workgroup tile.
// 8 waves in a 4(M) x 2(N) grid; each wave owns 32(M) x 64(N) = 2x4 fragments.
// As: [128 m][APITCH] row-major (m,k);  Bs: [128 n][BPITCH] transposed (n,k).
// A layout per ISA: lanes 0-15 -> M=lane, K 0..7 / 16..23; lanes 16-31 -> K 8..15 / 24..31.
// B layout: lane n in 0..15, lanes 0-15 hold K 0..7 / 16..23; lanes 16-31 -> K 8..15 / 24..31.
// ---------------------------------------------------------------------------
__device__ __forceinline__ void mma_step(const bf16* __restrict__ As,
                                         const bf16* __restrict__ Bs,
                                         int lane, int wm, int wn,
                                         v8f (&acc)[2][4])
{
  FragU a0, a1;
  const int mrow = wm * 32 + (lane & 15);
  const int kb   = (lane >> 4) * 8;
  a0.h[0] = *(const B16x8*)(As + mrow * APITCH + kb);
  a0.h[1] = *(const B16x8*)(As + mrow * APITCH + kb + 16);
  a1.h[0] = *(const B16x8*)(As + (mrow + 16) * APITCH + kb);
  a1.h[1] = *(const B16x8*)(As + (mrow + 16) * APITCH + kb + 16);
  const int kbb = (lane >> 4) * 8;
#pragma unroll
  for (int nf = 0; nf < 4; ++nf) {
    FragU b;
    const int n = wn * 64 + nf * 16 + (lane & 15);
    b.h[0] = *(const B16x8*)(Bs + n * BPITCH + kbb);
    b.h[1] = *(const B16x8*)(Bs + n * BPITCH + kbb + 16);
    acc[0][nf] = __builtin_amdgcn_wmma_f32_16x16x32_bf16(
        false, a0.v, false, b.v, (short)0, acc[0][nf], false, false);
    acc[1][nf] = __builtin_amdgcn_wmma_f32_16x16x32_bf16(
        false, a1.v, false, b.v, (short)0, acc[1][nf], false, false);
  }
}

// ---------------------------------------------------------------------------
// Mqk = Wq^T Wk  (64x64, K=64)
// ---------------------------------------------------------------------------
__global__ __launch_bounds__(256)
void k_mqk(const float* __restrict__ wq, const float* __restrict__ wk,
           float* __restrict__ Mqk)
{
  for (int idx = threadIdx.x; idx < 64 * 64; idx += 256) {
    const int d = idx >> 6, d2 = idx & 63;
    float s = 0.f;
    for (int o = 0; o < 64; ++o) s += wq[o * 64 + d] * wk[o * 64 + d2];
    Mqk[idx] = s;
  }
}

// ---------------------------------------------------------------------------
// bcomb[e] = fc2_b[e] + sum_m fc2_w[e,m] * fc1_b[m]
// ---------------------------------------------------------------------------
__global__ __launch_bounds__(256)
void k_bcomb(const float* __restrict__ W2, const float* __restrict__ b1,
             const float* __restrict__ b2, float* __restrict__ bc)
{
  const int e = blockIdx.x * 256 + threadIdx.x;   // 4 blocks -> 1024
  float s = 0.f;
  for (int m = 0; m < MAPD; ++m) s += W2[(size_t)e * MAPD + m] * b1[m];
  bc[e] = s + b2[e];
}

// ---------------------------------------------------------------------------
// Weight fusion GEMM: G[e',e] = sum_m fc1_w[m,e'] * fc2_w[e,m]   (bf16 out)
// M=e'(1024), N=e(1024), K=m(4096). 8x8 workgroups of 128x128 tiles.
// ---------------------------------------------------------------------------
__global__ __launch_bounds__(256)
void k_fuse_w(const float* __restrict__ W1, const float* __restrict__ W2,
              bf16* __restrict__ G)
{
  __shared__ bf16 As[128 * APITCH];
  __shared__ bf16 Bs[128 * BPITCH];
  const int tid = threadIdx.x, lane = tid & 31, w = tid >> 5;
  const int wm = w >> 1, wn = w & 1;
  const int m0 = blockIdx.y * 128;   // e' block
  const int n0 = blockIdx.x * 128;   // e  block
  const v8f z = {0.f, 0.f, 0.f, 0.f, 0.f, 0.f, 0.f, 0.f};
  v8f acc[2][4] = {{z, z, z, z}, {z, z, z, z}};

  for (int k0 = 0; k0 < MAPD; k0 += 32) {
#pragma unroll
    for (int it = 0; it < 16; ++it) {
      const int idx = tid + it * 256;            // 0..4095
      { // As[i][mm] = bf16(W1[k0+mm, m0+i])  (coalesced along e')
        const int i = idx & 127, mm = idx >> 7;
        As[i * APITCH + mm] = (bf16)W1[(size_t)(k0 + mm) * EMBED + m0 + i];
      }
      { // Bs[j][mm] = bf16(W2[n0+j, k0+mm])  (coalesced along m)
        const int mm = idx & 31, j = idx >> 5;
        Bs[j * BPITCH + mm] = (bf16)W2[(size_t)(n0 + j) * MAPD + k0 + mm];
      }
    }
    __syncthreads();
    mma_step(As, Bs, lane, wm, wn, acc);
    __syncthreads();
  }

#pragma unroll
  for (int mf = 0; mf < 2; ++mf)
#pragma unroll
    for (int nf = 0; nf < 4; ++nf) {
      const int n  = n0 + wn * 64 + nf * 16 + (lane & 15);
      const int mb = m0 + wm * 32 + mf * 16 + ((lane >> 4) << 3);
#pragma unroll
      for (int j = 0; j < 8; ++j)
        G[(size_t)(mb + j) * EMBED + n] = (bf16)acc[mf][nf][j];
    }
}

// ---------------------------------------------------------------------------
// Pass 1: diag scores via quadratic form with Mqk (in LDS) + column sums of inp.
// Block = (c, 16 t's); one (t,h) pair per thread.
// ---------------------------------------------------------------------------
__global__ __launch_bounds__(256)
void k_pass1(const float* __restrict__ inp, const float* __restrict__ Mqk,
             float* __restrict__ wsm, float* __restrict__ xsum)
{
  __shared__ float sM[64 * 64];
  const int tid = threadIdx.x;
  const int c = blockIdx.y;
  const int t0 = blockIdx.x * 16;
  for (int i = tid; i < 4096; i += 256) sM[i] = Mqk[i];
  __syncthreads();

  const int t = t0 + (tid >> 4);
  const int h = tid & 15;
  const float* xh = inp + (size_t)(c * TDIM + t) * EMBED + h * PER;
  float x[64];
#pragma unroll
  for (int d = 0; d < 64; ++d) x[d] = xh[d];
  float acc = 0.f;
  for (int d = 0; d < 64; ++d) {
    float tmp = 0.f;
#pragma unroll
    for (int d2 = 0; d2 < 64; ++d2) tmp += sM[d * 64 + d2] * x[d2];
    acc += x[d] * tmp;
  }
  wsm[(size_t)(c * TDIM + t) * HEADS + h] = acc * (1.0f / 32.0f);  // /sqrt(1024)

  // partial column sums over this block's 16 t's (L2-hot re-read)
  for (int e = tid; e < EMBED; e += 256) {
    float s = 0.f;
    for (int tt = 0; tt < 16; ++tt)
      s += inp[(size_t)(c * TDIM + t0 + tt) * EMBED + e];
    atomicAdd(&xsum[c * EMBED + e], s);
  }
}

// ---------------------------------------------------------------------------
// vsum[c, h*64+o] = sum_d wv[o,d] * xsum[c, h*64+d]
// ---------------------------------------------------------------------------
__global__ __launch_bounds__(256)
void k_vsum(const float* __restrict__ xsum, const float* __restrict__ wv,
            float* __restrict__ vs)
{
  const int c = blockIdx.x;
  for (int idx = threadIdx.x; idx < EMBED; idx += 256) {
    const int h = idx >> 6, o = idx & 63;
    float s = 0.f;
#pragma unroll 8
    for (int d = 0; d < 64; ++d) s += wv[o * 64 + d] * xsum[c * EMBED + h * 64 + d];
    vs[c * EMBED + idx] = s;
  }
}

// ---------------------------------------------------------------------------
// Softmax over C (32 values) per (t,h); in-place on wsm[c][t*16+h].
// ---------------------------------------------------------------------------
__global__ __launch_bounds__(256)
void k_softmax(float* __restrict__ wsm)
{
  const int gid = blockIdx.x * 256 + threadIdx.x;   // < 32768
  float v[32], m = -1e30f;
#pragma unroll
  for (int c = 0; c < 32; ++c) { v[c] = wsm[(size_t)c * 32768 + gid]; m = fmaxf(m, v[c]); }
  float s = 0.f;
#pragma unroll
  for (int c = 0; c < 32; ++c) { v[c] = __expf(v[c] - m); s += v[c]; }
  const float r = 1.f / s;
#pragma unroll
  for (int c = 0; c < 32; ++c) wsm[(size_t)c * 32768 + gid] = v[c] * r;
}

// ---------------------------------------------------------------------------
// Pass 2: x2 = inp + w*vsum; LN1; y = x2 + n1 -> bf16.  One row per block.
// ---------------------------------------------------------------------------
__global__ __launch_bounds__(256)
void k_pass2(const float* __restrict__ inp, const float* __restrict__ wsm,
             const float* __restrict__ vs, const float* __restrict__ g1,
             const float* __restrict__ b1, bf16* __restrict__ Y)
{
  __shared__ float sm1[8], sm2[8];
  const int r = blockIdx.x;          // c*2048 + t
  const int c = r >> 11;
  const int tid = threadIdx.x;
  const int e0 = tid * 4;
  const int h = e0 >> 6;

  const F4 xi = *(const F4*)(inp + (size_t)r * EMBED + e0);
  const float w = wsm[(size_t)r * HEADS + h];
  const F4 vv = *(const F4*)(vs + c * EMBED + e0);
  const float x0 = xi.x + w * vv.x, x1 = xi.y + w * vv.y;
  const float x2 = xi.z + w * vv.z, x3 = xi.w + w * vv.w;

  float s = x0 + x1 + x2 + x3;
  float q = x0 * x0 + x1 * x1 + x2 * x2 + x3 * x3;
#pragma unroll
  for (int o = 16; o; o >>= 1) { s += __shfl_xor(s, o, 32); q += __shfl_xor(q, o, 32); }
  if ((tid & 31) == 0) { sm1[tid >> 5] = s; sm2[tid >> 5] = q; }
  __syncthreads();
  float ts = 0.f, tq = 0.f;
#pragma unroll
  for (int i = 0; i < 8; ++i) { ts += sm1[i]; tq += sm2[i]; }
  const float mu = ts * (1.f / 1024.f);
  const float var = tq * (1.f / 1024.f) - mu * mu;
  const float rstd = rsqrtf(var + LNEPS);

  const F4 gg = *(const F4*)(g1 + e0);
  const F4 bb = *(const F4*)(b1 + e0);
  BF4 o4;
  o4.a = (bf16)(x0 + ((x0 - mu) * rstd * gg.x + bb.x));
  o4.b = (bf16)(x1 + ((x1 - mu) * rstd * gg.y + bb.y));
  o4.c = (bf16)(x2 + ((x2 - mu) * rstd * gg.z + bb.z));
  o4.d = (bf16)(x3 + ((x3 - mu) * rstd * gg.w + bb.w));
  *(BF4*)(Y + (size_t)r * EMBED + e0) = o4;
}

// ---------------------------------------------------------------------------
// Main GEMM: out_pre[r,e] = sum_{e'} Y[r,e'] * G[e',e] + bcomb[e]
// M=65536, N=1024, K=1024. 128x128 tiles, bf16 WMMA, f32 accum.
// Double-buffered LDS; A tiles staged with async global->LDS (ASYNCcnt),
// B tiles transposed through VGPRs (needs (n,k)-major layout).
// ---------------------------------------------------------------------------
__global__ __launch_bounds__(256)
void k_mlp_gemm(const bf16* __restrict__ Y, const bf16* __restrict__ G,
                const float* __restrict__ bias, float* __restrict__ out)
{
  __shared__ bf16 As[2][128 * APITCH];
  __shared__ bf16 Bs[2][128 * BPITCH];
  const int tid = threadIdx.x, lane = tid & 31, w = tid >> 5;
  const int wm = w >> 1, wn = w & 1;
  const int r0 = blockIdx.y * 128;
  const int n0 = blockIdx.x * 128;
  const v8f z = {0.f, 0.f, 0.f, 0.f, 0.f, 0.f, 0.f, 0.f};
  v8f acc[2][4] = {{z, z, z, z}, {z, z, z, z}};

  auto stage = [&](int buf, int k0) {
    // A: 128 rows x 32 k -> LDS directly, 16B async copies, coalesced
#pragma unroll
    for (int it = 0; it < 2; ++it) {
      const int idx = (tid + it * 256) * 8;
      const int row = idx >> 5, col = idx & 31;
      async_load_b128(&As[buf][row * APITCH + col],
                      Y + (size_t)(r0 + row) * EMBED + k0 + col);
    }
    // B: read G[k][n] coalesced along n, scatter (n,k)-major into LDS
#pragma unroll
    for (int it = 0; it < 2; ++it) {
      const int idx = (tid + it * 256) * 8;
      const int kk = idx >> 7, j = idx & 127;
      const B16x8 t = *(const B16x8*)(G + (size_t)(k0 + kk) * EMBED + n0 + j);
      const bf16* e = (const bf16*)&t;
#pragma unroll
      for (int qq = 0; qq < 8; ++qq) Bs[buf][(j + qq) * BPITCH + kk] = e[qq];
    }
    // prefetch next-next K tile of the (L2-resident) weight matrix
    if (k0 + 32 < EMBED)
      __builtin_prefetch(G + (size_t)(k0 + 32) * EMBED + n0 + tid * 4, 0, 1);
  };

  stage(0, 0);
  wait_asynccnt0();
  __syncthreads();

  for (int k0 = 0, i = 0; k0 < EMBED; k0 += 32, ++i) {
    const int cur = i & 1;
    if (k0 + 32 < EMBED) stage(cur ^ 1, k0 + 32);   // load next while computing
    mma_step(As[cur], Bs[cur], lane, wm, wn, acc);
    wait_asynccnt0();      // next tile's async A copies complete
    __syncthreads();       // + all waves done reading cur, B stores visible
  }

#pragma unroll
  for (int mf = 0; mf < 2; ++mf)
#pragma unroll
    for (int nf = 0; nf < 4; ++nf) {
      const int n = n0 + wn * 64 + nf * 16 + (lane & 15);
      const float bb = bias[n];
      const int mb = r0 + wm * 32 + mf * 16 + ((lane >> 4) << 3);
#pragma unroll
      for (int j = 0; j < 8; ++j)
        out[(size_t)(mb + j) * EMBED + n] = acc[mf][nf][j] + bb;
    }
}

// ---------------------------------------------------------------------------
// Final LayerNorm (in place on d_out). One row per block.
// ---------------------------------------------------------------------------
__global__ __launch_bounds__(256)
void k_ln2(float* __restrict__ out, const float* __restrict__ g,
           const float* __restrict__ b)
{
  __shared__ float sm1[8], sm2[8];
  const int r = blockIdx.x;
  const int tid = threadIdx.x;
  const int e0 = tid * 4;
  const F4 x = *(const F4*)(out + (size_t)r * EMBED + e0);
  float s = x.x + x.y + x.z + x.w;
  float q = x.x * x.x + x.y * x.y + x.z * x.z + x.w * x.w;
#pragma unroll
  for (int o = 16; o; o >>= 1) { s += __shfl_xor(s, o, 32); q += __shfl_xor(q, o, 32); }
  if ((tid & 31) == 0) { sm1[tid >> 5] = s; sm2[tid >> 5] = q; }
  __syncthreads();
  float ts = 0.f, tq = 0.f;
#pragma unroll
  for (int i = 0; i < 8; ++i) { ts += sm1[i]; tq += sm2[i]; }
  const float mu = ts * (1.f / 1024.f);
  const float var = tq * (1.f / 1024.f) - mu * mu;
  const float rstd = rsqrtf(var + LNEPS);
  const F4 gg = *(const F4*)(g + e0);
  const F4 bb = *(const F4*)(b + e0);
  F4 o;
  o.x = (x.x - mu) * rstd * gg.x + bb.x;
  o.y = (x.y - mu) * rstd * gg.y + bb.y;
  o.z = (x.z - mu) * rstd * gg.z + bb.z;
  o.w = (x.w - mu) * rstd * gg.w + bb.w;
  *(F4*)(out + (size_t)r * EMBED + e0) = o;
}

// ---------------------------------------------------------------------------
extern "C" void kernel_launch(void* const* d_in, const int* in_sizes, int n_in,
                              void* d_out, int out_size, void* d_ws, size_t ws_size,
                              hipStream_t stream)
{
  (void)in_sizes; (void)n_in; (void)out_size; (void)ws_size;
  const float* inp  = (const float*)d_in[0];
  const float* wq   = (const float*)d_in[1];
  const float* wk   = (const float*)d_in[2];
  const float* wvp  = (const float*)d_in[3];
  const float* ln1g = (const float*)d_in[4];
  const float* ln1b = (const float*)d_in[5];
  const float* fc1w = (const float*)d_in[6];
  const float* fc1b = (const float*)d_in[7];
  const float* fc2w = (const float*)d_in[8];
  const float* fc2b = (const float*)d_in[9];
  const float* ln2g = (const float*)d_in[10];
  const float* ln2b = (const float*)d_in[11];
  float* out = (float*)d_out;

  // workspace carve-up (~141 MB total)
  char* ws = (char*)d_ws;
  bf16*  Y    = (bf16*)ws;  ws += (size_t)ROWS * EMBED * 2;   // 128 MB  y (bf16)
  float* wsm  = (float*)ws; ws += (size_t)ROWS * HEADS * 4;   //   4 MB  diag/softmax
  float* xsum = (float*)ws; ws += (size_t)CDIM * EMBED * 4;   // 128 KB
  float* vsum = (float*)ws; ws += (size_t)CDIM * EMBED * 4;   // 128 KB
  float* mqk  = (float*)ws; ws += (size_t)64 * 64 * 4;        //  16 KB
  bf16*  G    = (bf16*)ws;  ws += (size_t)EMBED * EMBED * 2;  //   2 MB  fused weights
  float* bc   = (float*)ws; ws += (size_t)EMBED * 4;          //   4 KB  fused bias

  hipMemsetAsync(xsum, 0, (size_t)CDIM * EMBED * sizeof(float), stream);
  k_mqk   <<<1, 256, 0, stream>>>(wq, wk, mqk);
  k_bcomb <<<4, 256, 0, stream>>>(fc2w, fc1b, fc2b, bc);
  k_fuse_w<<<dim3(EMBED / 128, EMBED / 128), 256, 0, stream>>>(fc1w, fc2w, G);
  k_pass1 <<<dim3(TDIM / 16, CDIM), 256, 0, stream>>>(inp, mqk, wsm, xsum);
  k_vsum  <<<CDIM, 256, 0, stream>>>(xsum, wvp, vsum);
  k_softmax<<<(TDIM * HEADS) / 256, 256, 0, stream>>>(wsm);
  k_pass2 <<<ROWS, 256, 0, stream>>>(inp, wsm, vsum, ln1g, ln1b, Y);
  k_mlp_gemm<<<dim3(EMBED / 128, ROWS / 128), 256, 0, stream>>>(Y, G, bc, out);
  k_ln2   <<<ROWS, 256, 0, stream>>>(out, ln2g, ln2b);
}
